// HGGNN_35802847380150
// MI455X (gfx1250) — matrix operand order repaired and checked by reading, not compiled, not run on hardware
//
#include <hip/hip_runtime.h>
#include <hip/hip_bf16.h>
#include <math.h>

// Problem constants (match reference setup_inputs)
#define GN 50000
#define GE 800000

typedef __attribute__((ext_vector_type(16))) __bf16 v16bf;
typedef __attribute__((ext_vector_type(8)))  float  v8f;

union FragAB { uint4 q[2]; v16bf v; };

__device__ __forceinline__ unsigned pack_bf2(float a, float b) {
  // round-to-nearest-even fp32 -> bf16, packed pair
  unsigned ua = __float_as_uint(a); ua = ua + 0x7FFFu + ((ua >> 16) & 1u);
  unsigned ub = __float_as_uint(b); ub = ub + 0x7FFFu + ((ub >> 16) & 1u);
  return (ua >> 16) | (ub & 0xFFFF0000u);
}

__device__ __forceinline__ void atomAddF(float* p, float v) {
  __hip_atomic_fetch_add(p, v, __ATOMIC_RELAXED, __HIP_MEMORY_SCOPE_AGENT);
}

// ---------------------------------------------------------------------------
// Fused dual-output WMMA GEMM:
//   C1[M x 128] = A[M x K] * B1[K x 128]
//   C2[M x 128] = A[M x K] * B2[K x 128]
// A/B fp32 in memory, converted to bf16 while staging into LDS; fp32
// accumulation via v_wmma_f32_16x16x32_bf16. Block tile 128x128 (both
// outputs), 8 waves (4x2), each wave computes a 32x64 sub-tile per output
// = 2 x (2x4) WMMA 16x16 accumulators. Sharing the A tile across the two
// outputs halves HBM traffic for the activations.
// ---------------------------------------------------------------------------
__global__ __launch_bounds__(256)
void k_gemm_dual_bf16wmma(const float* __restrict__ A,
                          const float* __restrict__ B1,
                          const float* __restrict__ B2,
                          float* __restrict__ C1, float* __restrict__ C2,
                          int M, int K) {
  // A tile: 128 rows x 32 halves, row stride 20 uints (40 halves, padded)
  // B tiles: 32 rows x 128 halves, row stride 68 uints (136 halves, padded)
  __shared__ __align__(16) unsigned Alds[128 * 20];
  __shared__ __align__(16) unsigned Blds[2][32 * 68];

  const int t    = threadIdx.x;
  const int lane = t & 31;
  const int w    = t >> 5;
  const int mr   = w >> 1;   // 0..3 : wave row group (32 rows each)
  const int nc   = w & 1;    // 0..1 : wave col group (64 cols each)
  const int row0 = blockIdx.x * 128;

  v8f acc[2][2][4];          // [output][mi][ni]
#pragma unroll
  for (int o = 0; o < 2; ++o)
#pragma unroll
    for (int mi = 0; mi < 2; ++mi)
#pragma unroll
      for (int ni = 0; ni < 4; ++ni)
        acc[o][mi][ni] = (v8f){0.f, 0.f, 0.f, 0.f, 0.f, 0.f, 0.f, 0.f};

  const int ar = t >> 1, acol = (t & 1) * 16;   // A staging: row, 16-float chunk
  const int br = t >> 3, bcol = (t & 7) * 16;   // B staging: row, 16-float chunk

  for (int kc = 0; kc < K; kc += 32) {
    // ---- stage A chunk (128 x 32 fp32 -> bf16 in LDS), shared by both ----
    {
      const int rg = row0 + ar;
      unsigned* dst = &Alds[ar * 20 + (t & 1) * 8];
      if (rg < M) {
        const float4* ap = (const float4*)(A + (size_t)rg * K + kc + acol);
        if (kc + 32 < K)
          __builtin_prefetch(A + (size_t)rg * K + kc + 32 + acol, 0, 0);
        float4 f0 = ap[0], f1 = ap[1], f2 = ap[2], f3 = ap[3];
        dst[0] = pack_bf2(f0.x, f0.y); dst[1] = pack_bf2(f0.z, f0.w);
        dst[2] = pack_bf2(f1.x, f1.y); dst[3] = pack_bf2(f1.z, f1.w);
        dst[4] = pack_bf2(f2.x, f2.y); dst[5] = pack_bf2(f2.z, f2.w);
        dst[6] = pack_bf2(f3.x, f3.y); dst[7] = pack_bf2(f3.z, f3.w);
      } else {
#pragma unroll
        for (int q = 0; q < 8; ++q) dst[q] = 0u;
      }
    }
    // ---- stage both B chunks (32 x 128 fp32 -> bf16 in LDS) ----
    {
      const size_t boff = (size_t)(kc + br) * 128 + bcol;
      const float4* bp1 = (const float4*)(B1 + boff);
      const float4* bp2 = (const float4*)(B2 + boff);
      unsigned* d1 = &Blds[0][br * 68 + (t & 7) * 8];
      unsigned* d2 = &Blds[1][br * 68 + (t & 7) * 8];
      float4 f0 = bp1[0], f1 = bp1[1], f2 = bp1[2], f3 = bp1[3];
      d1[0] = pack_bf2(f0.x, f0.y); d1[1] = pack_bf2(f0.z, f0.w);
      d1[2] = pack_bf2(f1.x, f1.y); d1[3] = pack_bf2(f1.z, f1.w);
      d1[4] = pack_bf2(f2.x, f2.y); d1[5] = pack_bf2(f2.z, f2.w);
      d1[6] = pack_bf2(f3.x, f3.y); d1[7] = pack_bf2(f3.z, f3.w);
      f0 = bp2[0]; f1 = bp2[1]; f2 = bp2[2]; f3 = bp2[3];
      d2[0] = pack_bf2(f0.x, f0.y); d2[1] = pack_bf2(f0.z, f0.w);
      d2[2] = pack_bf2(f1.x, f1.y); d2[3] = pack_bf2(f1.z, f1.w);
      d2[4] = pack_bf2(f2.x, f2.y); d2[5] = pack_bf2(f2.z, f2.w);
      d2[6] = pack_bf2(f3.x, f3.y); d2[7] = pack_bf2(f3.z, f3.w);
    }
    __syncthreads();

    // ---- build A fragments once, then per-output B fragments + WMMAs ----
    FragAB af[2];
    const int lm  = lane & 15;
    const int kbU = (lane & 16) ? 4 : 0;  // lanes 0-15: K 0-7/16-23; 16-31: K 8-15/24-31
#pragma unroll
    for (int mi = 0; mi < 2; ++mi) {
      const int m = mr * 32 + mi * 16 + lm;
      af[mi].q[0] = *(const uint4*)&Alds[m * 20 + kbU];
      af[mi].q[1] = *(const uint4*)&Alds[m * 20 + kbU + 8];
    }
#pragma unroll
    for (int o = 0; o < 2; ++o) {
      FragAB bfr[4];
#pragma unroll
      for (int ni = 0; ni < 4; ++ni) {
        const int cN = nc * 64 + ni * 16 + lm;  // lane = N column, elements = K (table order)
        const __bf16* p = (const __bf16*)&Blds[o][0];
#pragma unroll
        for (int s = 0; s < 16; ++s) {
          const int kk = (s < 8) ? (2 * kbU + s) : (16 + 2 * kbU + (s - 8));
          bfr[ni].v[s] = p[kk * 136 + cN];
        }
      }
#pragma unroll
      for (int mi = 0; mi < 2; ++mi)
#pragma unroll
        for (int ni = 0; ni < 4; ++ni)
          acc[o][mi][ni] = __builtin_amdgcn_wmma_f32_16x16x32_bf16(
              false, af[mi].v, false, bfr[ni].v, (short)0, acc[o][mi][ni],
              false, false);
    }
    __syncthreads();
  }

  // ---- store C1/C2 (fp32); C/D layout: lane = N, VGPR r = M (0-7 / 8-15) ----
  const int cn = lane & 15;
  const int rb = (lane >> 4) * 8;
#pragma unroll
  for (int mi = 0; mi < 2; ++mi)
#pragma unroll
    for (int ni = 0; ni < 4; ++ni) {
      const int col = nc * 64 + ni * 16 + cn;
#pragma unroll
      for (int r = 0; r < 8; ++r) {
        const int row = row0 + mr * 32 + mi * 16 + rb + r;
        if (row < M) {
          C1[(size_t)row * 128 + col] = acc[0][mi][ni][r];
          C2[(size_t)row * 128 + col] = acc[1][mi][ni][r];
        }
      }
    }
}

// ---------------------------------------------------------------------------
// Per-node attention logits: s_src[n,h] = sum_f proj[n,h,f]*a_src[h,f], etc.
// ---------------------------------------------------------------------------
__global__ void k_scores(const float* __restrict__ proj,
                         const float* __restrict__ asrc,
                         const float* __restrict__ atrg,
                         float* __restrict__ ss, float* __restrict__ st, int n2) {
  const int i = blockIdx.x * blockDim.x + threadIdx.x;
  if (i >= n2) return;
  const int node = i >> 1, h = i & 1;
  const float* p  = proj + (size_t)node * 128 + h * 64;
  const float* as = asrc + h * 64;
  const float* at = atrg + h * 64;
  float a = 0.f, b = 0.f;
#pragma unroll 8
  for (int f = 0; f < 64; ++f) {
    const float v = p[f];
    a += v * as[f];
    b += v * at[f];
  }
  ss[i] = a;
  st[i] = b;
}

// Zero segment-sum denominators, reset global-max scalar (orderable encoding)
__global__ void k_init(float* __restrict__ denom, unsigned* __restrict__ gmax, int n2) {
  const int i = blockIdx.x * blockDim.x + threadIdx.x;
  if (i < n2) denom[i] = 0.f;
  if (i == 0) *gmax = 0u;  // orderable encoding of minimum
}

// Per-edge leaky-relu score + global max (orderable-uint atomic max)
__global__ void k_edge_score_max(const int* __restrict__ src, const int* __restrict__ trg,
                                 const float* __restrict__ ss, const float* __restrict__ st,
                                 float* __restrict__ escore, unsigned* __restrict__ gmax,
                                 int e2) {
  const int i = blockIdx.x * blockDim.x + threadIdx.x;
  unsigned enc = 0u;
  if (i < e2) {
    const int e = i >> 1, h = i & 1;
    const int s = src[e], tg = trg[e];
    float v = ss[s * 2 + h] + st[tg * 2 + h];
    v = (v > 0.f) ? v : 0.2f * v;  // leaky_relu, slope 0.2
    escore[i] = v;
    const unsigned u = __float_as_uint(v);
    enc = (u & 0x80000000u) ? ~u : (u | 0x80000000u);
  }
  __shared__ unsigned sm[256];
  sm[threadIdx.x] = enc;
  __syncthreads();
  for (int s2 = 128; s2 > 0; s2 >>= 1) {
    if (threadIdx.x < (unsigned)s2) {
      const unsigned o = sm[threadIdx.x + s2];
      if (o > sm[threadIdx.x]) sm[threadIdx.x] = o;
    }
    __syncthreads();
  }
  if (threadIdx.x == 0)
    __hip_atomic_fetch_max(gmax, sm[0], __ATOMIC_RELAXED, __HIP_MEMORY_SCOPE_AGENT);
}

// exp(score - gmax) in place + segment-sum denominator via fp32 atomics
__global__ void k_edge_exp(const int* __restrict__ trg, float* __restrict__ escore,
                           float* __restrict__ denom, const unsigned* __restrict__ gmax,
                           int e2) {
  const int i = blockIdx.x * blockDim.x + threadIdx.x;
  if (i >= e2) return;
  const unsigned enc = *gmax;
  const unsigned u = (enc & 0x80000000u) ? (enc ^ 0x80000000u) : ~enc;
  const float m = __uint_as_float(u);
  const int e = i >> 1, h = i & 1;
  const float v = expf(escore[i] - m);
  escore[i] = v;
  atomAddF(&denom[trg[e] * 2 + h], v);
}

// Weighted message scatter: agg[trg] += proj[src] * att  (128 floats/edge)
__global__ void k_scatter(const int* __restrict__ src, const int* __restrict__ trg,
                          const float* __restrict__ proj, const float* __restrict__ escore,
                          const float* __restrict__ denom, float* __restrict__ agg,
                          long long total) {
  const long long stride = (long long)gridDim.x * blockDim.x;
  for (long long i = (long long)blockIdx.x * blockDim.x + threadIdx.x; i < total; i += stride) {
    const int e = (int)(i >> 7);
    const int j = (int)(i & 127);
    const int h = j >> 6;
    const int s = src[e], tg = trg[e];
    const float att = escore[(size_t)e * 2 + h] / (denom[tg * 2 + h] + 1e-16f);
    const float val = proj[(size_t)s * 128 + j] * att;
    atomAddF(&agg[(size_t)tg * 128 + j], val);
  }
}

// Layer-0 epilogue: h = relu(elu(agg + b0))   (agg already contains skip-GEMM)
__global__ void k_finish0(const float* __restrict__ agg, const float* __restrict__ b0,
                          float* __restrict__ hbuf, int total) {
  const int i = blockIdx.x * blockDim.x + threadIdx.x;
  if (i >= total) return;
  const float z = agg[i] + b0[i & 127];
  const float e = (z > 0.f) ? z : (expf(z) - 1.f);
  hbuf[i] = fmaxf(e, 0.f);
}

// Layer-2 epilogue: out = mean over heads + b2
__global__ void k_final(const float* __restrict__ agg, const float* __restrict__ b2,
                        float* __restrict__ out, int total) {
  const int i = blockIdx.x * blockDim.x + threadIdx.x;
  if (i >= total) return;
  const int n = i >> 6, f = i & 63;
  out[i] = 0.5f * (agg[(size_t)n * 128 + f] + agg[(size_t)n * 128 + 64 + f]) + b2[f];
}

// ---------------------------------------------------------------------------
extern "C" void kernel_launch(void* const* d_in, const int* in_sizes, int n_in,
                              void* d_out, int out_size, void* d_ws, size_t ws_size,
                              hipStream_t stream) {
  (void)in_sizes; (void)n_in; (void)out_size; (void)ws_size;
  const int N = GN, E = GE;

  const float* x      = (const float*)d_in[0];
  const int*   ei     = (const int*)d_in[1];
  const float* W0     = (const float*)d_in[2];
  const float* asrc0  = (const float*)d_in[3];
  const float* atrg0  = (const float*)d_in[4];
  const float* Wskip0 = (const float*)d_in[5];
  const float* b0     = (const float*)d_in[6];
  const float* W2     = (const float*)d_in[7];
  const float* asrc2  = (const float*)d_in[8];
  const float* atrg2  = (const float*)d_in[9];
  const float* Wskip2 = (const float*)d_in[10];
  const float* b2     = (const float*)d_in[11];

  const int* src = ei;       // edge_index[0]
  const int* trg = ei + E;   // edge_index[1]

  // Workspace carve-up (fp32 units); total ~84.4 MB
  float* proj   = (float*)d_ws;
  float* agg    = proj + (size_t)N * 128;
  float* hbuf   = agg  + (size_t)N * 128;
  float* ssrc   = hbuf + (size_t)N * 128;
  float* strg   = ssrc + (size_t)N * 2;
  float* escore = strg + (size_t)N * 2;
  float* denom  = escore + (size_t)E * 2;
  unsigned* gmax = (unsigned*)(denom + (size_t)N * 2);

  float* out = (float*)d_out;

  const int TB = 256;
  const dim3 gGemm((N + 127) / 128);
  const int gN2   = (N * 2 + TB - 1) / TB;
  const int gE2   = (E * 2 + TB - 1) / TB;
  const int gNF   = (N * 128 + TB - 1) / TB;
  const int gOut  = (N * 64 + TB - 1) / TB;
  const long long scatTotal = (long long)E * 128;

  // ======================= Layer 0 (concat=True, elu, then relu) ============
  k_gemm_dual_bf16wmma<<<gGemm, TB, 0, stream>>>(x, W0, Wskip0, proj, agg, N, 256);
  k_scores<<<gN2, TB, 0, stream>>>(proj, asrc0, atrg0, ssrc, strg, N * 2);
  k_init<<<gN2, TB, 0, stream>>>(denom, gmax, N * 2);
  k_edge_score_max<<<gE2, TB, 0, stream>>>(src, trg, ssrc, strg, escore, gmax, E * 2);
  k_edge_exp<<<gE2, TB, 0, stream>>>(trg, escore, denom, gmax, E * 2);
  k_scatter<<<8192, TB, 0, stream>>>(src, trg, proj, escore, denom, agg, scatTotal);
  k_finish0<<<gNF, TB, 0, stream>>>(agg, b0, hbuf, N * 128);

  // ======================= Layer 2 (concat=False, identity act) =============
  k_gemm_dual_bf16wmma<<<gGemm, TB, 0, stream>>>(hbuf, W2, Wskip2, proj, agg, N, 128);
  k_scores<<<gN2, TB, 0, stream>>>(proj, asrc2, atrg2, ssrc, strg, N * 2);
  k_init<<<gN2, TB, 0, stream>>>(denom, gmax, N * 2);
  k_edge_score_max<<<gE2, TB, 0, stream>>>(src, trg, ssrc, strg, escore, gmax, E * 2);
  k_edge_exp<<<gE2, TB, 0, stream>>>(trg, escore, denom, gmax, E * 2);
  k_scatter<<<8192, TB, 0, stream>>>(src, trg, proj, escore, denom, agg, scatTotal);
  k_final<<<gOut, TB, 0, stream>>>(agg, b2, out, N * 64);
}
